// ShapeBottleneckModel_66932770341113
// MI455X (gfx1250) — matrix operand order repaired and checked by reading, not compiled, not run on hardware
//
#include <hip/hip_runtime.h>

typedef __attribute__((ext_vector_type(16))) _Float16 v16h;
typedef __attribute__((ext_vector_type(8)))  float    v8f;

#define BATCH 32
#define CHAN  8
#define TLEN  512
#define NSH   5
// output layout (floats): out[320] | dists[5120] | probs[5120] | loss[1]
#define DIST_OFF 320
#define PROB_OFF 5440
#define LOSS_OFF 10560

// ---------------------------------------------------------------------------
// Kernel 1: per-(b,c) row normalization, ddof=1 std. One block per row.
// ---------------------------------------------------------------------------
__global__ void sbm_normalize(const float* __restrict__ x, float* __restrict__ xn) {
    __shared__ float red[256];
    const int row = blockIdx.x;           // b*8 + c
    const int tid = threadIdx.x;
    const float* xr = x + (size_t)row * TLEN;
    __builtin_prefetch(xr + tid, 0, 1);   // global_prefetch_b8 (gfx1250)

    float v0 = xr[tid];
    float v1 = xr[tid + 256];

    // mean
    red[tid] = v0 + v1;
    __syncthreads();
    for (int st = 128; st > 0; st >>= 1) {
        if (tid < st) red[tid] += red[tid + st];
        __syncthreads();
    }
    const float mu = red[0] * (1.0f / TLEN);
    __syncthreads();

    // variance (ddof=1)
    float d0 = v0 - mu, d1 = v1 - mu;
    red[tid] = d0 * d0 + d1 * d1;
    __syncthreads();
    for (int st = 128; st > 0; st >>= 1) {
        if (tid < st) red[tid] += red[tid + st];
        __syncthreads();
    }
    const float sd  = sqrtf(red[0] * (1.0f / (TLEN - 1)));
    const float inv = 1.0f / (sd + 1e-8f);

    xn[(size_t)row * TLEN + tid]       = d0 * inv;
    xn[(size_t)row * TLEN + tid + 256] = d1 * inv;
}

// ---------------------------------------------------------------------------
// Kernel 2: sliding L1 shapelet distances + min/max reductions over m.
// One block per (b,c) row; x-row, w-row, pcm-row staged in LDS.
// Writes dists and probs directly into d_out.
// ---------------------------------------------------------------------------
__global__ void sbm_shapelet(const float* __restrict__ xn,
                             const float* __restrict__ w0, const float* __restrict__ w1,
                             const float* __restrict__ w2, const float* __restrict__ w3,
                             const float* __restrict__ p0, const float* __restrict__ p1,
                             const float* __restrict__ p2, const float* __restrict__ p3,
                             float* __restrict__ out) {
    __shared__ float s_x[TLEN];
    __shared__ float s_w[256];
    __shared__ float s_pcm[461];
    __shared__ float red_min[256];
    __shared__ float red_max[256];

    const int Ls[4]   = {52, 103, 154, 256};
    const int Ms[4]   = {461, 410, 359, 257};
    const int offs[4] = {0, 40, 80, 120};
    const float* ws[4]   = {w0, w1, w2, w3};
    const float* pcms[4] = {p0, p1, p2, p3};

    const int row = blockIdx.x;
    const int bi  = row >> 3;
    const int ci  = row & 7;
    const int tid = threadIdx.x;

    for (int i = tid; i < TLEN; i += 256) s_x[i] = xn[(size_t)row * TLEN + i];

    for (int s = 0; s < 4; ++s) {
        const int L = Ls[s], M = Ms[s];
        const float invL = 1.0f / (float)L;
        __syncthreads();                       // previous set done with s_pcm
        for (int i = tid; i < M; i += 256) s_pcm[i] = pcms[s][ci * M + i];

        for (int n = 0; n < NSH; ++n) {
            __syncthreads();                   // previous use of s_w done
            for (int i = tid; i < L; i += 256) s_w[i] = ws[s][((size_t)(n * CHAN + ci)) * L + i];
            __syncthreads();

            float dmin = 3.4e38f, pmax = -3.4e38f;
            for (int m = tid; m < M; m += 256) {
                float acc = 0.0f;
                #pragma unroll 4
                for (int l = 0; l < L; ++l) acc += fabsf(s_x[m + l] - s_w[l]);
                const float ds = acc * invL * s_pcm[m];
                const float p  = expf(-(ds * ds));   // EPS == 1
                dmin = fminf(dmin, ds);
                pmax = fmaxf(pmax, p);
            }
            red_min[tid] = dmin;
            red_max[tid] = pmax;
            __syncthreads();
            for (int st = 128; st > 0; st >>= 1) {
                if (tid < st) {
                    red_min[tid] = fminf(red_min[tid], red_min[tid + st]);
                    red_max[tid] = fmaxf(red_max[tid], red_max[tid + st]);
                }
                __syncthreads();
            }
            if (tid == 0) {
                const int col = offs[s] + n * CHAN + ci;
                out[DIST_OFF + bi * 160 + col] = red_min[0];
                out[PROB_OFF + bi * 160 + col] = red_max[0];
            }
        }
    }
}

// ---------------------------------------------------------------------------
// Kernel 3: out(32,10) = probs(32,160) @ W_out(10,160)^T via v_wmma_f32_16x16x32_f16.
// Single 32-thread wave (EXEC all-1s). 2 M-tiles x 5 K-steps, f32 accumulate.
// Fragment layouts per CDNA5 ISA 7.12.2.
// ---------------------------------------------------------------------------
__global__ void sbm_gemm_wmma(const float* __restrict__ probs,
                              const float* __restrict__ Wout,
                              float* __restrict__ out) {
    __shared__ _Float16 sA[32 * 160];   // probs, row-major
    __shared__ _Float16 sB[16 * 160];   // W_out padded to 16 rows, row-major
    const int tid = threadIdx.x;        // 32 threads == 1 wave

    for (int i = tid; i < 32 * 160; i += 32) sA[i] = (_Float16)probs[i];
    for (int i = tid; i < 16 * 160; i += 32) {
        const int nrow = i / 160, k = i % 160;
        sB[i] = (nrow < 10) ? (_Float16)Wout[nrow * 160 + k] : (_Float16)0.0f;
    }
    __syncthreads();

    const int lane = tid & 31;
    const int half = lane >> 4;   // 0: lanes 0-15, 1: lanes 16-31
    const int lrow = lane & 15;

    for (int mt = 0; mt < 2; ++mt) {
        v8f c = {0.f, 0.f, 0.f, 0.f, 0.f, 0.f, 0.f, 0.f};
        #pragma unroll
        for (int kk = 0; kk < 5; ++kk) {
            const int kbase = kk * 32;
            v16h a, b;
            // A fragment: 16x32 f16. lanes 0-15: M=row; VGPR0-3 -> K=8*half+2v+w,
            // VGPR4-7 -> K=16+8*half+2(v-4)+w
            #pragma unroll
            for (int i = 0; i < 16; ++i) {
                const int v = i >> 1, wb = i & 1;
                const int kl = (v < 4) ? (8 * half + 2 * v + wb)
                                       : (16 + 8 * half + 2 * (v - 4) + wb);
                a[i] = sA[(mt * 16 + lrow) * 160 + kbase + kl];
            }
            // B fragment: 32x16 f16. N = lane%16; K = 8*half + i (i<8), 16 + 8*half + (i-8) (i>=8)
            #pragma unroll
            for (int i = 0; i < 16; ++i) {
                const int kb = (i < 8) ? (8 * half + i) : (16 + 8 * half + (i - 8));
                b[i] = sB[lrow * 160 + kbase + kb];
            }
            c = __builtin_amdgcn_wmma_f32_16x16x32_f16(
                    /*neg_a=*/false, a, /*neg_b=*/false, b,
                    /*c_mod=*/(short)0, c, /*reuse_a=*/false, /*reuse_b=*/false);
        }
        // D layout: VGPR r holds M = r + 8*half, N = lane%16
        #pragma unroll
        for (int r = 0; r < 8; ++r) {
            const int mrow = mt * 16 + r + 8 * half;
            const int ncol = lrow;
            if (ncol < 10) out[mrow * 10 + ncol] = c[r];
        }
    }
}

// ---------------------------------------------------------------------------
// Kernel 4: scalar loss = 0.1*mean|W_out| + 0.1*sum_s diversity(w_s)
// ---------------------------------------------------------------------------
__global__ void sbm_loss(const float* __restrict__ w0, const float* __restrict__ w1,
                         const float* __restrict__ w2, const float* __restrict__ w3,
                         const float* __restrict__ Wout, float* __restrict__ out) {
    __shared__ float red[256];
    const int tid = threadIdx.x;
    const int Ls[4] = {52, 103, 154, 256};
    const float* ws[4] = {w0, w1, w2, w3};

    float absW = 0.0f;
    for (int i = tid; i < 10 * 160; i += 256) absW += fabsf(Wout[i]);

    // 4 sets * 8 chans * 20 ordered off-diagonal pairs = 640 pairs
    float div = 0.0f;
    for (int pid = tid; pid < 640; pid += 256) {
        const int s  = pid / 160;
        const int r  = pid % 160;
        const int ci = r / 20;
        const int pr = r % 20;
        const int ai = pr / 4;
        const int j0 = pr % 4;
        const int bi = j0 + ((j0 >= ai) ? 1 : 0);
        const int L = Ls[s];
        const float* wa = ws[s] + ((size_t)(ai * CHAN + ci)) * L;
        const float* wb = ws[s] + ((size_t)(bi * CHAN + ci)) * L;
        float ss = 0.0f;
        for (int l = 0; l < L; ++l) {
            const float d = wb[l] - wa[l] + 1e-6f;   // diff[c,a,b,l] = sh[c,b]-sh[c,a]+eps
            ss += d * d;
        }
        div += expf(-sqrtf(ss)) * (1.0f / 200.0f);   // mean over c*n*n = 200 per set
    }

    red[tid] = absW;
    __syncthreads();
    for (int st = 128; st > 0; st >>= 1) {
        if (tid < st) red[tid] += red[tid + st];
        __syncthreads();
    }
    const float absW_tot = red[0];
    __syncthreads();
    red[tid] = div;
    __syncthreads();
    for (int st = 128; st > 0; st >>= 1) {
        if (tid < st) red[tid] += red[tid + st];
        __syncthreads();
    }
    if (tid == 0) {
        out[LOSS_OFF] = 0.1f * (absW_tot / 1600.0f) + 0.1f * red[0];
    }
}

// ---------------------------------------------------------------------------
extern "C" void kernel_launch(void* const* d_in, const int* in_sizes, int n_in,
                              void* d_out, int out_size, void* d_ws, size_t ws_size,
                              hipStream_t stream) {
    // dict order: x, w0, pcm0, w1, pcm1, w2, pcm2, w3, pcm3, W_out
    const float* x    = (const float*)d_in[0];
    const float* w0   = (const float*)d_in[1];
    const float* pcm0 = (const float*)d_in[2];
    const float* w1   = (const float*)d_in[3];
    const float* pcm1 = (const float*)d_in[4];
    const float* w2   = (const float*)d_in[5];
    const float* pcm2 = (const float*)d_in[6];
    const float* w3   = (const float*)d_in[7];
    const float* pcm3 = (const float*)d_in[8];
    const float* Wout = (const float*)d_in[9];
    float* out = (float*)d_out;
    float* xn  = (float*)d_ws;               // 32*8*512 f32 = 512 KB scratch

    sbm_normalize<<<BATCH * CHAN, 256, 0, stream>>>(x, xn);
    sbm_shapelet <<<BATCH * CHAN, 256, 0, stream>>>(xn, w0, w1, w2, w3,
                                                    pcm0, pcm1, pcm2, pcm3, out);
    sbm_gemm_wmma<<<1, 32, 0, stream>>>(out + PROB_OFF, Wout, out);
    sbm_loss     <<<1, 256, 0, stream>>>(w0, w1, w2, w3, Wout, out);
}
